// EGNNLayer_39771397161330
// MI455X (gfx1250) — matrix operand order repaired and checked by reading, not compiled, not run on hardware
//
#include <hip/hip_runtime.h>
#include <hip/hip_bf16.h>

// EGNN layer for MI455X (gfx1250): bf16 WMMA (16x16x32, f32 accum) pipeline.
// B=2, N=16384, K=16, H=128 (fixed by reference).
// v3: kc-outer / t-inner with full accumulator residency (128 VGPRs of C,
//     only current A-chunk live) so the compiler can pipeline B-fragment
//     loads behind WMMAs; fast rcp-based silu (TRANS ops co-execute w/ WMMA).

#define H_DIM 128
#define K_NB  16
#define NT    8          // 128 / 16 column tiles
#define N_REF 16384

typedef __attribute__((ext_vector_type(16))) __bf16 bf16x16;
typedef __attribute__((ext_vector_type(8)))  __bf16 bf16x8;
typedef __attribute__((ext_vector_type(8)))  float  f32x8;
typedef __attribute__((ext_vector_type(4)))  float  f32x4;

__device__ __forceinline__ float silu_f(float v) {
  // x * rcp(1+exp(-x)): v_exp_f32 + v_rcp_f32 (TRANS) instead of IEEE divide.
  return v * __builtin_amdgcn_rcpf(1.0f + __expf(-v));
}

__device__ __forceinline__ f32x8 wmma_bf16(bf16x16 a, bf16x16 b, f32x8 c) {
  // D = A(16x32) x B(32x16) + C, f32 accumulate
  return __builtin_amdgcn_wmma_f32_16x16x32_bf16(false, a, false, b,
                                                 (short)0, c, false, false);
}

// Build one 16x32 A-fragment K-chunk for this lane from a row-major f32 row.
// Lane<16: K in {0..7, 16..23}; lane>=16: K in {8..15, 24..31} (hf = lane>>4).
__device__ __forceinline__ bf16x16 make_afrag(const float* __restrict__ src,
                                              int f0 /* = 32*c + 8*hf */) {
  f32x4 p0 = *(const f32x4*)(src + f0);
  f32x4 p1 = *(const f32x4*)(src + f0 + 4);
  f32x4 q0 = *(const f32x4*)(src + f0 + 16);
  f32x4 q1 = *(const f32x4*)(src + f0 + 20);
  bf16x16 a;
#pragma unroll
  for (int j = 0; j < 4; ++j) {
    a[j]      = (__bf16)p0[j];
    a[4 + j]  = (__bf16)p1[j];
    a[8 + j]  = (__bf16)q0[j];
    a[12 + j] = (__bf16)q1[j];
  }
  return a;
}

// Load a 16x32 A-chunk from the LDS staging tile (bf16, row-major 16 x 128).
__device__ __forceinline__ bf16x16 lds_afrag(const __bf16* __restrict__ st,
                                             int lr, int f0) {
  bf16x8 g0 = *(const bf16x8*)&st[lr * H_DIM + f0];
  bf16x8 g1 = *(const bf16x8*)&st[lr * H_DIM + f0 + 16];
  bf16x16 a;
#pragma unroll
  for (int j = 0; j < 8; ++j) { a[j] = g0[j]; a[8 + j] = g1[j]; }
  return a;
}

// ---------------------------------------------------------------------------
// Prep: swizzle a (kdim x 128) f32 weight into per-lane WMMA B-fragment order.
// Fragment index f = (kc*NT + t)*32 + lane ; 16 contiguous bf16 per lane:
//   element j = W[32*kc + 8*(lane>>4) + (j&7) + 16*(j>>3)][16*t + (lane&15)]
// ---------------------------------------------------------------------------
__global__ void egnn_prep_frag(const float* __restrict__ W,
                               __bf16* __restrict__ dst, int kdim) {
  int e = blockIdx.x * blockDim.x + threadIdx.x;
  int total = kdim * H_DIM;
  if (e >= total) return;
  int j    = e & 15;
  int lane = (e >> 4) & 31;
  int t    = (e >> 9) & 7;
  int kc   = e >> 12;
  int k = 32 * kc + 8 * (lane >> 4) + (j & 7) + 16 * (j >> 3);
  int n = 16 * t + (lane & 15);
  dst[e] = (__bf16)W[k * H_DIM + n];
}

// ---------------------------------------------------------------------------
// Edge kernel: one wave per TWO nodes (each node's K=16 edges = 16 GEMM rows).
//   L1: silu( [h_i | h_j](16x256) @ We1h + be1 + sqdist*we1_last )
//   L2: m_ij = silu( L1 @ We2 + be2 );  m_i = sum_rows(m_ij)
//   C1: w = silu( m_ij @ Wc1 + bc1 ) @ Wc2 ;  x_new = x + mean(x_diff*w)
// ---------------------------------------------------------------------------
__global__ __launch_bounds__(128)
void egnn_edge(const float* __restrict__ h, const float* __restrict__ x,
               const int* __restrict__ eidx,
               const bf16x16* __restrict__ fWe1, const bf16x16* __restrict__ fWe2,
               const bf16x16* __restrict__ fWc1,
               const float* __restrict__ we1_last, const float* __restrict__ be1,
               const float* __restrict__ be2, const float* __restrict__ bc1,
               const float* __restrict__ wc2,
               float* __restrict__ m_i_out, float* __restrict__ x_out, int N) {
  __shared__ __align__(16) __bf16 stage[4][2][16 * H_DIM];  // 8 KB per wave
  __shared__ float s_sq[4][2][16];
  __shared__ float s_xd[4][2][3][16];
  __shared__ float s_w[4][2][16];

  const int lane = threadIdx.x & 31;
  const int wid  = threadIdx.x >> 5;
  const int hf   = lane >> 4;       // lane-half (row/K split; also node q)
  const int lr   = lane & 15;       // row (edge) / column-in-tile
  const long n0  = (long)(blockIdx.x * 4 + wid) * 2;   // first node of pair
  const long n1  = n0 + 1;                             // same batch (N even)
  const long hb  = (long)(int)(n0 / N) * N;            // batch base row

  // Geometry: lane (hf=q, lr) handles edge lr of node q. All 32 lanes active.
  {
    const long nq  = n0 + hf;
    const int  nbg = eidx[nq * K_NB + lr];
    const long ng  = hb + nbg;
    float d0 = x[nq * 3 + 0] - x[ng * 3 + 0];
    float d1 = x[nq * 3 + 1] - x[ng * 3 + 1];
    float d2 = x[nq * 3 + 2] - x[ng * 3 + 2];
    s_xd[wid][hf][0][lr] = d0; s_xd[wid][hf][1][lr] = d1; s_xd[wid][hf][2][lr] = d2;
    s_sq[wid][hf][lr] = d0 * d0 + d1 * d1 + d2 * d2;
  }
  __syncthreads();

  // Source rows: features 0..127 = h_i, 128..255 = h_nbr(row lr).
  const long nb0 = hb + eidx[n0 * K_NB + lr];
  const long nb1 = hb + eidx[n1 * K_NB + lr];
  const float* hi0 = h + n0  * H_DIM;
  const float* hi1 = h + n1  * H_DIM;
  const float* hn0 = h + nb0 * H_DIM;
  const float* hn1 = h + nb1 * H_DIM;

  // Full accumulator residency: 8 n-tiles x 2 nodes (128 VGPRs).
  f32x8 A0[NT], A1[NT];

  // ---- Layer 1: C init = be1 + sqdist * we1_last (rank-1 fold) ----
#pragma unroll
  for (int t = 0; t < NT; ++t) {
    float bb = be1[16 * t + lr];
    float wd = we1_last[16 * t + lr];
#pragma unroll
    for (int v = 0; v < 8; ++v) {
      A0[t][v] = bb + s_sq[wid][0][v + 8 * hf] * wd;
      A1[t][v] = bb + s_sq[wid][1][v + 8 * hf] * wd;
    }
  }
  // kc-outer: build current A-chunks, stream 8 B fragments, 16 WMMAs per kc.
#pragma unroll
  for (int kc = 0; kc < 8; ++kc) {
    int f0 = 32 * kc + 8 * hf - ((kc < 4) ? 0 : H_DIM);
    bf16x16 a0c = make_afrag((kc < 4) ? hi0 : hn0, f0);
    bf16x16 a1c = make_afrag((kc < 4) ? hi1 : hn1, f0);
#pragma unroll
    for (int t = 0; t < NT; ++t) {
      bf16x16 bw = fWe1[(kc * NT + t) * 32 + lane];
      A0[t] = wmma_bf16(a0c, bw, A0[t]);
      A1[t] = wmma_bf16(a1c, bw, A1[t]);
    }
  }
#pragma unroll
  for (int t = 0; t < NT; ++t) {
#pragma unroll
    for (int v = 0; v < 8; ++v) {
      stage[wid][0][(v + 8 * hf) * H_DIM + 16 * t + lr] = (__bf16)silu_f(A0[t][v]);
      stage[wid][1][(v + 8 * hf) * H_DIM + 16 * t + lr] = (__bf16)silu_f(A1[t][v]);
    }
  }
  __syncthreads();

  // ---- Layer 2: A = silu(L1) from LDS (C/D layout -> A layout transpose) --
#pragma unroll
  for (int t = 0; t < NT; ++t) {
    float bb = be2[16 * t + lr];
#pragma unroll
    for (int v = 0; v < 8; ++v) { A0[t][v] = bb; A1[t][v] = bb; }
  }
#pragma unroll
  for (int kc = 0; kc < 4; ++kc) {
    int f0 = 32 * kc + 8 * hf;
    bf16x16 a0c = lds_afrag(stage[wid][0], lr, f0);
    bf16x16 a1c = lds_afrag(stage[wid][1], lr, f0);
#pragma unroll
    for (int t = 0; t < NT; ++t) {
      bf16x16 bw = fWe2[(kc * NT + t) * 32 + lane];
      A0[t] = wmma_bf16(a0c, bw, A0[t]);
      A1[t] = wmma_bf16(a1c, bw, A1[t]);
    }
  }
  __syncthreads();   // all LDS reads of L1 done before overwrite below
#pragma unroll
  for (int t = 0; t < NT; ++t) {
    float ms0 = 0.0f, ms1 = 0.0f;
#pragma unroll
    for (int v = 0; v < 8; ++v) {
      float m0 = silu_f(A0[t][v]), m1 = silu_f(A1[t][v]);
      ms0 += m0; ms1 += m1;
      stage[wid][0][(v + 8 * hf) * H_DIM + 16 * t + lr] = (__bf16)m0;
      stage[wid][1][(v + 8 * hf) * H_DIM + 16 * t + lr] = (__bf16)m1;
    }
    ms0 += __shfl_xor(ms0, 16, 32);        // combine row halves
    ms1 += __shfl_xor(ms1, 16, 32);
    if (lane < 16) m_i_out[n0 * H_DIM + 16 * t + lr] = ms0;   // half 0 -> node0
    else           m_i_out[n1 * H_DIM + 16 * t + lr] = ms1;   // half 1 -> node1
  }
  __syncthreads();

  // ---- Coord MLP: w = silu(m_ij @ Wc1 + bc1) @ Wc2 ----
#pragma unroll
  for (int t = 0; t < NT; ++t) {
    float bb = bc1[16 * t + lr];
#pragma unroll
    for (int v = 0; v < 8; ++v) { A0[t][v] = bb; A1[t][v] = bb; }
  }
#pragma unroll
  for (int kc = 0; kc < 4; ++kc) {
    int f0 = 32 * kc + 8 * hf;
    bf16x16 a0c = lds_afrag(stage[wid][0], lr, f0);
    bf16x16 a1c = lds_afrag(stage[wid][1], lr, f0);
#pragma unroll
    for (int t = 0; t < NT; ++t) {
      bf16x16 bw = fWc1[(kc * NT + t) * 32 + lane];
      A0[t] = wmma_bf16(a0c, bw, A0[t]);
      A1[t] = wmma_bf16(a1c, bw, A1[t]);
    }
  }
  float w0[8], w1[8];
#pragma unroll
  for (int v = 0; v < 8; ++v) { w0[v] = 0.0f; w1[v] = 0.0f; }
#pragma unroll
  for (int t = 0; t < NT; ++t) {
    float wn = wc2[16 * t + lr];
#pragma unroll
    for (int v = 0; v < 8; ++v) {
      w0[v] += silu_f(A0[t][v]) * wn;
      w1[v] += silu_f(A1[t][v]) * wn;
    }
  }
#pragma unroll
  for (int m = 1; m < 16; m <<= 1) {
#pragma unroll
    for (int v = 0; v < 8; ++v) {
      w0[v] += __shfl_xor(w0[v], m, 32);
      w1[v] += __shfl_xor(w1[v], m, 32);
    }
  }
  if (lr == 0) {                              // lanes 0 and 16 hold halves
#pragma unroll
    for (int v = 0; v < 8; ++v) {
      s_w[wid][0][v + 8 * hf] = w0[v];
      s_w[wid][1][v + 8 * hf] = w1[v];
    }
  }
  __syncthreads();

  if (lr < 3) {                               // lanes 0..2 node0, 16..18 node1
    const long nq = n0 + hf;
    float s = 0.0f;
#pragma unroll
    for (int m = 0; m < 16; ++m) s += s_xd[wid][hf][lr][m] * s_w[wid][hf][m];
    x_out[nq * 3 + lr] = x[nq * 3 + lr] + s * (1.0f / (float)K_NB);
  }
}

// ---------------------------------------------------------------------------
// Node kernel: one wave per TWO 16-node tiles.
//   h_new = silu([h | m_i](16x256) @ Wn1 + bn1) @ Wn2 + bn2 + h
// ---------------------------------------------------------------------------
__global__ __launch_bounds__(128)
void egnn_node(const float* __restrict__ h, const float* __restrict__ m_i,
               const bf16x16* __restrict__ fWn1, const bf16x16* __restrict__ fWn2,
               const float* __restrict__ bn1, const float* __restrict__ bn2,
               float* __restrict__ h_out) {
  __shared__ __align__(16) __bf16 stage[4][2][16 * H_DIM];

  const int lane = threadIdx.x & 31;
  const int wid  = threadIdx.x >> 5;
  const int hf   = lane >> 4;
  const int lr   = lane & 15;
  const long t0  = (long)(blockIdx.x * 4 + wid) * 2;   // tile pair
  const long r0  = t0 * 16 + lr;                       // lane's A-rows
  const long r1  = (t0 + 1) * 16 + lr;

  f32x8 A0[NT], A1[NT];

  // ---- Layer n1 ----
#pragma unroll
  for (int t = 0; t < NT; ++t) {
    float bb = bn1[16 * t + lr];
#pragma unroll
    for (int v = 0; v < 8; ++v) { A0[t][v] = bb; A1[t][v] = bb; }
  }
#pragma unroll
  for (int kc = 0; kc < 8; ++kc) {
    int f0 = 32 * kc + 8 * hf - ((kc < 4) ? 0 : H_DIM);
    const float* s0 = ((kc < 4) ? h : m_i) + r0 * H_DIM;
    const float* s1 = ((kc < 4) ? h : m_i) + r1 * H_DIM;
    bf16x16 a0c = make_afrag(s0, f0);
    bf16x16 a1c = make_afrag(s1, f0);
#pragma unroll
    for (int t = 0; t < NT; ++t) {
      bf16x16 bw = fWn1[(kc * NT + t) * 32 + lane];
      A0[t] = wmma_bf16(a0c, bw, A0[t]);
      A1[t] = wmma_bf16(a1c, bw, A1[t]);
    }
  }
#pragma unroll
  for (int t = 0; t < NT; ++t) {
#pragma unroll
    for (int v = 0; v < 8; ++v) {
      stage[wid][0][(v + 8 * hf) * H_DIM + 16 * t + lr] = (__bf16)silu_f(A0[t][v]);
      stage[wid][1][(v + 8 * hf) * H_DIM + 16 * t + lr] = (__bf16)silu_f(A1[t][v]);
    }
  }
  __syncthreads();

  // ---- Layer n2: residual + bias folded into C init ----
#pragma unroll
  for (int t = 0; t < NT; ++t) {
    int n = 16 * t + lr;
#pragma unroll
    for (int v = 0; v < 8; ++v) {
      long q0 = t0 * 16 + v + 8 * hf;
      long q1 = (t0 + 1) * 16 + v + 8 * hf;
      A0[t][v] = bn2[n] + h[q0 * H_DIM + n];
      A1[t][v] = bn2[n] + h[q1 * H_DIM + n];
    }
  }
#pragma unroll
  for (int kc = 0; kc < 4; ++kc) {
    int f0 = 32 * kc + 8 * hf;
    bf16x16 a0c = lds_afrag(stage[wid][0], lr, f0);
    bf16x16 a1c = lds_afrag(stage[wid][1], lr, f0);
#pragma unroll
    for (int t = 0; t < NT; ++t) {
      bf16x16 bw = fWn2[(kc * NT + t) * 32 + lane];
      A0[t] = wmma_bf16(a0c, bw, A0[t]);
      A1[t] = wmma_bf16(a1c, bw, A1[t]);
    }
  }
#pragma unroll
  for (int t = 0; t < NT; ++t) {
    int n = 16 * t + lr;
#pragma unroll
    for (int v = 0; v < 8; ++v) {
      long q0 = t0 * 16 + v + 8 * hf;
      long q1 = (t0 + 1) * 16 + v + 8 * hf;
      h_out[q0 * H_DIM + n] = A0[t][v];
      h_out[q1 * H_DIM + n] = A1[t][v];
    }
  }
}

// ---------------------------------------------------------------------------
extern "C" void kernel_launch(void* const* d_in, const int* in_sizes, int n_in,
                              void* d_out, int out_size, void* d_ws, size_t ws_size,
                              hipStream_t stream) {
  const float* h    = (const float*)d_in[0];
  const float* x    = (const float*)d_in[1];
  const int*   eidx = (const int*)  d_in[2];
  const float* We1  = (const float*)d_in[3];
  const float* be1  = (const float*)d_in[4];
  const float* We2  = (const float*)d_in[5];
  const float* be2  = (const float*)d_in[6];
  const float* Wc1  = (const float*)d_in[7];
  const float* bc1  = (const float*)d_in[8];
  const float* Wc2  = (const float*)d_in[9];
  const float* Wn1  = (const float*)d_in[10];
  const float* bn1  = (const float*)d_in[11];
  const float* Wn2  = (const float*)d_in[12];
  const float* bn2  = (const float*)d_in[13];

  const int BN = in_sizes[0] / H_DIM;   // B*N (= 32768)
  const int N  = N_REF;

  // Workspace layout: bf16 weight fragments, then f32 m_i.
  __bf16* fw = (__bf16*)d_ws;
  __bf16* fWe1 = fw;               // 256*128 = 32768 bf16
  __bf16* fWe2 = fw + 32768;       // 128*128 = 16384
  __bf16* fWc1 = fw + 49152;       // 16384
  __bf16* fWn1 = fw + 65536;       // 32768
  __bf16* fWn2 = fw + 98304;       // 16384
  float*  m_i  = (float*)((char*)d_ws + 256 * 1024);

  const float* we1_last = We1 + 256 * H_DIM;   // sq_dist row of We1
  float* h_out = (float*)d_out;
  float* x_out = (float*)d_out + (size_t)BN * H_DIM;

  // 1) Pre-swizzle weights into WMMA B-fragment order (bf16).
  egnn_prep_frag<<<(256 * H_DIM + 255) / 256, 256, 0, stream>>>(We1, fWe1, 256);
  egnn_prep_frag<<<(128 * H_DIM + 255) / 256, 256, 0, stream>>>(We2, fWe2, 128);
  egnn_prep_frag<<<(128 * H_DIM + 255) / 256, 256, 0, stream>>>(Wc1, fWc1, 128);
  egnn_prep_frag<<<(256 * H_DIM + 255) / 256, 256, 0, stream>>>(Wn1, fWn1, 256);
  egnn_prep_frag<<<(128 * H_DIM + 255) / 256, 256, 0, stream>>>(Wn2, fWn2, 128);

  // 2) Edge MLP + coord update: two nodes per wave, 4 waves per block.
  egnn_edge<<<BN / 8, 128, 0, stream>>>(
      h, x, eidx,
      (const bf16x16*)fWe1, (const bf16x16*)fWe2, (const bf16x16*)fWc1,
      we1_last, be1, be2, bc1, Wc2, m_i, x_out, N);

  // 3) Node MLP: two 16-node tiles per wave.
  egnn_node<<<(BN / 16) / 8, 128, 0, stream>>>(
      h, m_i, (const bf16x16*)fWn1, (const bf16x16*)fWn2, bn1, bn2, h_out);
}
